// NT_Xent_MSE_68882685493998
// MI455X (gfx1250) — matrix operand order repaired and checked by reading, not compiled, not run on hardware
//
#include <hip/hip_runtime.h>
#include <hip/hip_bf16.h>

typedef __attribute__((ext_vector_type(16))) _Float16 v16h;
typedef __attribute__((ext_vector_type(8)))  _Float16 v8h;
typedef __attribute__((ext_vector_type(8)))  float    v8f;

#define N_TOT   8192          // 2*B
#define DIM     1024
#define BHALF   4096
#define INV_SC  (1.0f/512.0f) // 1/(D*TEMP) = 1/(1024*0.5)
#define SPLITS  8
#define M_WAVE  32            // rows per wave (2 row tiles)
#define NEG_BIG (-1e30f)

// ---------------------------------------------------------------------------
// Kernel 1: fp32 -> f16 conversion of z = concat(z_i, z_j), plus row ||z||^2
// ---------------------------------------------------------------------------
__global__ __launch_bounds__(256)
void prep_kernel(const float* __restrict__ zi, const float* __restrict__ zj,
                 _Float16* __restrict__ zh, float* __restrict__ sq) {
  __shared__ float red[256];
  const int row = blockIdx.x;
  const float* src = (row < BHALF) ? (zi + (size_t)row * DIM)
                                   : (zj + (size_t)(row - BHALF) * DIM);
  _Float16* dst = zh + (size_t)row * DIM;
  float acc = 0.f;
  for (int i = threadIdx.x; i < DIM; i += 256) {
    float v = src[i];
    dst[i] = (_Float16)v;
    acc = fmaf(v, v, acc);
  }
  red[threadIdx.x] = acc;
  __syncthreads();
  for (int off = 128; off > 0; off >>= 1) {
    if (threadIdx.x < (unsigned)off) red[threadIdx.x] += red[threadIdx.x + off];
    __syncthreads();
  }
  if (threadIdx.x == 0) sq[row] = red[0];
}

// ---------------------------------------------------------------------------
// Kernel 2: fused WMMA GEMM (z z^T) + online logsumexp epilogue.
// Each wave: 32-row stripe (2 row tiles) x 1/8 of the 8192 columns,
// 2x4 register blocking -> 8 WMMAs per 12 b128 loads per k-step.
// ---------------------------------------------------------------------------
__global__ __launch_bounds__(256)
void simlse_kernel(const _Float16* __restrict__ zh, const float* __restrict__ sq,
                   float* __restrict__ posv, float* __restrict__ mpart,
                   float* __restrict__ spart) {
  const int lane   = threadIdx.x & 31;
  const int waveId = threadIdx.x >> 5;
  const int gw     = blockIdx.x * 8 + waveId;    // global wave id
  const int stripe = gw / SPLITS;                // 0..255 (32-row stripe)
  const int split  = gw % SPLITS;                // column-range split
  const int mw     = stripe * M_WAVE;
  const int half   = lane >> 4;                  // lane-half (K-chunk select)
  const int l16    = lane & 15;

  // A fragments: row (mw + rt*16 + l16); K chunks at half*8 and half*8+16
  const _Float16* aBase[2];
  float sqrow[2][8];
  #pragma unroll
  for (int rt = 0; rt < 2; ++rt) {
    aBase[rt] = zh + (size_t)(mw + rt * 16 + l16) * DIM + half * 8;
    #pragma unroll
    for (int v = 0; v < 8; ++v) sqrow[rt][v] = sq[mw + rt * 16 + v + 8 * half];
  }

  // C layout: slot (lane, v) is row v+8*half, col (tileBase + l16).
  // Diagonal / positive slots satisfy l16 == v + 8*half.
  const int  diagv     = l16 - 8 * half;
  const bool slotOwner = (diagv >= 0) && (diagv < 8);

  float mrun[2][8], srun[2][8];
  #pragma unroll
  for (int rt = 0; rt < 2; ++rt)
    #pragma unroll
    for (int v = 0; v < 8; ++v) { mrun[rt][v] = NEG_BIG; srun[rt][v] = 0.f; }

  const int n0_beg = split * (N_TOT / SPLITS);
  const int n0_end = n0_beg + (N_TOT / SPLITS);
  const v8f zerov  = {};

  for (int n0 = n0_beg; n0 < n0_end; n0 += 64) {   // 4 col-tiles per iter
    v8f acc[2][4];
    const _Float16* bBase[4];
    float sqcol[4];
    #pragma unroll
    for (int cb = 0; cb < 4; ++cb) {
      acc[0][cb] = zerov;
      acc[1][cb] = zerov;
      const int col = n0 + cb * 16 + l16;
      // B fragment: column = l16; K chunks at half*8 and half*8+16
      bBase[cb] = zh + (size_t)col * DIM + half * 8;
      sqcol[cb] = sq[col];
    }

    #pragma unroll 2
    for (int k = 0; k < DIM; k += 32) {
      union { v16h v; v8h h[2]; } af[2];
      #pragma unroll
      for (int rt = 0; rt < 2; ++rt) {
        af[rt].h[0] = *(const v8h*)(aBase[rt] + k);
        af[rt].h[1] = *(const v8h*)(aBase[rt] + k + 16);
      }
      #pragma unroll
      for (int cb = 0; cb < 4; ++cb) {
        union { v16h v; v8h h[2]; } bf;
        bf.h[0] = *(const v8h*)(bBase[cb] + k);
        bf.h[1] = *(const v8h*)(bBase[cb] + k + 16);
        #pragma unroll
        for (int rt = 0; rt < 2; ++rt) {
          acc[rt][cb] = __builtin_amdgcn_wmma_f32_16x16x32_f16(
              /*neg_a=*/false, af[rt].v, /*neg_b=*/false, bf.v,
              /*c_mod=*/(short)0, acc[rt][cb],
              /*reuse_a=*/false, /*reuse_b=*/false);
        }
      }
    }

    // Epilogue: sim = (sq_i + sq_j - 2*S)/(D*TEMP), online (m,s) update
    #pragma unroll
    for (int rt = 0; rt < 2; ++rt) {
      const int rowBase = mw + rt * 16;
      const int posBase = (rowBase + BHALF) & (N_TOT - 1);
      #pragma unroll
      for (int cb = 0; cb < 4; ++cb) {
        const int  colBase    = n0 + cb * 16;
        const bool isDiagTile = (colBase == rowBase);
        const bool isPosTile  = (colBase == posBase);
        #pragma unroll
        for (int v = 0; v < 8; ++v) {
          const float s = acc[rt][cb][v];
          float x = fmaf(-2.0f * INV_SC, s, (sqrow[rt][v] + sqcol[cb]) * INV_SC);
          const bool mine = slotOwner && (diagv == v);
          if (isPosTile && mine) posv[rowBase + v + 8 * half] = x;  // positive
          if (isDiagTile && mine) x = NEG_BIG;                      // mask self
          const float nm = fmaxf(mrun[rt][v], x);
          srun[rt][v] = srun[rt][v] * __expf(mrun[rt][v] - nm) + __expf(x - nm);
          mrun[rt][v] = nm;
        }
      }
    }
  }

  // reduce (m,s) across the 16 lanes of each half (columns of each row)
  #pragma unroll
  for (int rt = 0; rt < 2; ++rt) {
    #pragma unroll
    for (int v = 0; v < 8; ++v) {
      float m = mrun[rt][v], s = srun[rt][v];
      #pragma unroll
      for (int off = 1; off < 16; off <<= 1) {
        const float mo = __shfl_xor(m, off, 32);
        const float so = __shfl_xor(s, off, 32);
        const float nm = fmaxf(m, mo);
        s = s * __expf(m - nm) + so * __expf(mo - nm);
        m = nm;
      }
      mrun[rt][v] = m; srun[rt][v] = s;
    }
    if (l16 == 0) {
      #pragma unroll
      for (int v = 0; v < 8; ++v) {
        const int row = mw + rt * 16 + v + 8 * half;
        mpart[row * SPLITS + split] = mrun[rt][v];
        spart[row * SPLITS + split] = srun[rt][v];
      }
    }
  }
}

// ---------------------------------------------------------------------------
// Kernel 3: deterministic cross-split LSE merge + scalar loss
// ---------------------------------------------------------------------------
__global__ __launch_bounds__(256)
void finish_kernel(const float* __restrict__ mpart, const float* __restrict__ spart,
                   const float* __restrict__ posv, float* __restrict__ out) {
  __shared__ float red[256];
  float acc = 0.f;
  for (int row = threadIdx.x; row < N_TOT; row += 256) {
    float m = NEG_BIG;
    #pragma unroll
    for (int p = 0; p < SPLITS; ++p) m = fmaxf(m, mpart[row * SPLITS + p]);
    float s = 0.f;
    #pragma unroll
    for (int p = 0; p < SPLITS; ++p)
      s += spart[row * SPLITS + p] * __expf(mpart[row * SPLITS + p] - m);
    acc += (m + __logf(s)) - posv[row];
  }
  red[threadIdx.x] = acc;
  __syncthreads();
  for (int off = 128; off > 0; off >>= 1) {
    if (threadIdx.x < (unsigned)off) red[threadIdx.x] += red[threadIdx.x + off];
    __syncthreads();
  }
  if (threadIdx.x == 0) out[0] = red[0] / (float)N_TOT;
}

// ---------------------------------------------------------------------------
extern "C" void kernel_launch(void* const* d_in, const int* in_sizes, int n_in,
                              void* d_out, int out_size, void* d_ws, size_t ws_size,
                              hipStream_t stream) {
  const float* zi = (const float*)d_in[0];
  const float* zj = (const float*)d_in[1];

  char* ws = (char*)d_ws;
  _Float16* zh  = (_Float16*)ws;                              // 16 MB
  float*    sq  = (float*)(ws + (size_t)N_TOT * DIM * 2);     // 32 KB
  float*    posv  = sq + N_TOT;                               // 32 KB
  float*    mpart = posv + N_TOT;                             // 256 KB
  float*    spart = mpart + (size_t)N_TOT * SPLITS;           // 256 KB

  prep_kernel<<<N_TOT, 256, 0, stream>>>(zi, zj, zh, sq);
  // 256 stripes * 8 splits = 2048... -> (8192/32)*8 = 2048 waves? no:
  // 256 stripes * 8 splits = 2048 waves / 8 per block = 256 blocks
  simlse_kernel<<<(N_TOT / M_WAVE) * SPLITS / 8, 256, 0, stream>>>(zh, sq, posv, mpart, spart);
  finish_kernel<<<1, 256, 0, stream>>>(mpart, spart, posv, (float*)d_out);
}
